// SelfAttention_21655225106765
// MI455X (gfx1250) — matrix operand order repaired and checked by reading, not compiled, not run on hardware
//
#include <hip/hip_runtime.h>
#include <math.h>

// ---- problem constants ----
#define BB 64
#define TT 256
#define CC 1024
#define HH 16
#define DH 64
#define MM (BB*TT)          // 16384 tokens
#define SCALE_F 0.125f      // 1/sqrt(64)

// ---- WMMA vector types (CDNA5 wave32) ----
typedef __attribute__((ext_vector_type(16))) __bf16 v16bf;
typedef __attribute__((ext_vector_type(8)))  __bf16 v8bf;
typedef __attribute__((ext_vector_type(8)))  float  v8f;

union BFrag16 { v16bf v; v8bf h[2]; };

__device__ __forceinline__ v8f wmma_bf16(v16bf a, v16bf b, v8f c) {
  // D = A(16x32 bf16) * B(32x16 bf16) + C(16x16 f32)
  return __builtin_amdgcn_wmma_f32_16x16x32_bf16(false, a, false, b,
                                                 (short)0, c, false, false);
}

__device__ __forceinline__ unsigned short f32_to_bf16(float f) {
  unsigned int u = __float_as_uint(f);
  u += 0x7FFFu + ((u >> 16) & 1u);      // round-to-nearest-even
  return (unsigned short)(u >> 16);
}

// Generic pointers into __shared__ carry the wave-relative LDS byte offset in
// their low 32 bits (ISA 10.2).
__device__ __forceinline__ unsigned to_lds_off(const unsigned short* p) {
  return (unsigned)(unsigned long long)p;
}

// CDNA5 async copy: global -> LDS, 16B per lane, tracked by ASYNCcnt.
// GVS mode: mem_addr = SADDR(sgpr64) + VADDR(vgpr32) ; VDST = LDS byte addr.
__device__ __forceinline__ void async_ld_b128(unsigned lds_addr,
                                              const unsigned short* sbase,
                                              unsigned byte_off) {
  asm volatile("global_load_async_to_lds_b128 %0, %1, %2"
               :: "v"(lds_addr), "v"(byte_off), "s"(sbase)
               : "memory");
}

// A fragment: 16x32 tile, row-major source (row stride in shorts).
// Lane L holds row L%16; K = koff..koff+15, koff = 16*(L/16).
__device__ __forceinline__ v16bf frag_a(const unsigned short* base, int stride) {
  const int lane = threadIdx.x & 31;
  const unsigned short* p = base + (lane & 15) * stride + ((lane >> 4) << 4);
  BFrag16 f;
  f.h[0] = *(const v8bf*)(p);
  f.h[1] = *(const v8bf*)(p + 8);
  return f.v;
}

// B fragment: 32x16 tile stored K-contiguous per column (source is [N][K]
// row-major, row stride in shorts). Lane L holds column L%16;
// K = koff..koff+15, koff = 16*(L/16).
__device__ __forceinline__ v16bf frag_b(const unsigned short* base, int stride) {
  const int lane = threadIdx.x & 31;
  const unsigned short* p = base + (lane & 15) * stride + ((lane >> 4) << 4);
  BFrag16 f;
  f.h[0] = *(const v8bf*)(p);
  f.h[1] = *(const v8bf*)(p + 8);
  return f.v;
}

// ---------------- kernel 1a: f32 -> bf16 convert ----------------
__global__ void __launch_bounds__(256) cvt_bf16_kernel(
    const float* __restrict__ src, unsigned short* __restrict__ dst, int n) {
  int i = blockIdx.x * blockDim.x + threadIdx.x;
  int stride = gridDim.x * blockDim.x;
  for (; i < n; i += stride) dst[i] = f32_to_bf16(src[i]);
}

// ---------------- kernel 1b: weight transpose + convert ----------------
// Wt[n][k] = W[k][n], both 1024x1024
__global__ void __launch_bounds__(256) transpose_bf16_kernel(
    const float* __restrict__ W, unsigned short* __restrict__ Wt) {
  int idx = blockIdx.x * blockDim.x + threadIdx.x;
  if (idx < CC * CC) {
    int k = idx >> 10;
    int n = idx & 1023;
    Wt[n * CC + k] = f32_to_bf16(W[idx]);
  }
}

// ---------------- async double-buffered GEMM mainloop pieces ----------------
#define LDSTRIDE 40                      // 32 + 8 shorts pad
#define BUFSH (128 * LDSTRIDE)           // shorts per LDS buffer
#define BUFBYTES (BUFSH * 2)             // bytes per LDS buffer

// Stage one 128x32-short tile (rows row0.., cols k0..k0+31) into LDS via
// async copies; 2 x b128 per thread (256 threads, 512 chunks of 16B).
__device__ __forceinline__ void stage_tile_async(unsigned ldsbase,
                                                 const unsigned short* g,
                                                 int row0, int k0) {
  const int tid = threadIdx.x;
#pragma unroll
  for (int i = 0; i < 2; ++i) {
    int chunk = tid + i * 256;
    int row = chunk >> 2;
    int col = (chunk & 3) << 3;
    unsigned goff = (unsigned)(((row0 + row) * CC + k0 + col) * 2);
    unsigned loff = ldsbase + (unsigned)((row * LDSTRIDE + col) * 2);
    async_ld_b128(loff, g, goff);
  }
}

// ---------------- kernel 2: fused QKV GEMM ----------------
// Y = xb[16384x1024] * W^T-layout[1024x1024]; z selects Q/K/V.
// Q,K stored [B,H,T,Dh]; V stored transposed [B,H,Dh,T].
__global__ void __launch_bounds__(256) qkv_gemm_kernel(
    const unsigned short* __restrict__ xb,
    const unsigned short* __restrict__ wtq,
    const unsigned short* __restrict__ wtk,
    const unsigned short* __restrict__ wtv,
    unsigned short* __restrict__ qo,
    unsigned short* __restrict__ ko,
    unsigned short* __restrict__ vto) {
  __shared__ __align__(16) unsigned short ldsA[2 * BUFSH];
  __shared__ __align__(16) unsigned short ldsB[2 * BUFSH];
  const int tid = threadIdx.x;
  const int wave = tid >> 5;
  const int wm = wave >> 1;     // 0..3 -> 32-row strip
  const int wn = wave & 1;      // 0..1 -> 64-col strip
  const int M0 = blockIdx.x * 128;
  const int N0 = blockIdx.y * 128;
  const int z = blockIdx.z;
  const unsigned short* wt = (z == 0) ? wtq : (z == 1) ? wtk : wtv;
  const unsigned ldsAo = to_lds_off(ldsA);
  const unsigned ldsBo = to_lds_off(ldsB);

  v8f zero = {};
  v8f acc[2][4];
#pragma unroll
  for (int mi = 0; mi < 2; ++mi)
#pragma unroll
    for (int ni = 0; ni < 4; ++ni) acc[mi][ni] = zero;

  // prologue: stage K-step 0 into buffer 0
  stage_tile_async(ldsAo, xb, M0, 0);
  stage_tile_async(ldsBo, wt, N0, 0);

  for (int it = 0; it < CC / 32; ++it) {
    const int p = it & 1;
    if (it + 1 < CC / 32) {
      // stage next K-step into the other buffer, then wait for current stage
      stage_tile_async(ldsAo + (1 - p) * BUFBYTES, xb, M0, (it + 1) * 32);
      stage_tile_async(ldsBo + (1 - p) * BUFBYTES, wt, N0, (it + 1) * 32);
      asm volatile("s_wait_asynccnt 0x4" ::: "memory");
    } else {
      asm volatile("s_wait_asynccnt 0x0" ::: "memory");
    }
    __syncthreads();
    const unsigned short* bufA = ldsA + p * BUFSH;
    const unsigned short* bufB = ldsB + p * BUFSH;
    v16bf aF[2], bF[4];
#pragma unroll
    for (int mi = 0; mi < 2; ++mi)
      aF[mi] = frag_a(bufA + (wm * 32 + mi * 16) * LDSTRIDE, LDSTRIDE);
#pragma unroll
    for (int ni = 0; ni < 4; ++ni)
      bF[ni] = frag_b(bufB + (wn * 64 + ni * 16) * LDSTRIDE, LDSTRIDE);
#pragma unroll
    for (int mi = 0; mi < 2; ++mi)
#pragma unroll
      for (int ni = 0; ni < 4; ++ni)
        acc[mi][ni] = wmma_bf16(aF[mi], bF[ni], acc[mi][ni]);
    __syncthreads();   // all waves done reading buf p before it is re-staged
  }

  const int lane = tid & 31, ln = lane & 15, lh = lane >> 4;
#pragma unroll
  for (int mi = 0; mi < 2; ++mi)
#pragma unroll
    for (int ni = 0; ni < 4; ++ni)
#pragma unroll
      for (int v = 0; v < 8; ++v) {
        int gm = M0 + wm * 32 + mi * 16 + v + (lh << 3);
        int gn = N0 + wn * 64 + ni * 16 + ln;
        unsigned short hb = f32_to_bf16(acc[mi][ni][v]);
        int b = gm >> 8, t = gm & 255, h = gn >> 6, d = gn & 63;
        size_t head = (size_t)(b * HH + h);
        if (z == 2)      vto[(head * DH + d) * TT + t] = hb;  // V transposed
        else if (z == 0) qo[(head * TT + t) * DH + d] = hb;
        else             ko[(head * TT + t) * DH + d] = hb;
      }
}

// ---------------- kernel 3: causal flash attention ----------------
// One workgroup per (b,h); wave w owns queries [32w, 32w+32).
__global__ void __launch_bounds__(256) attention_kernel(
    const unsigned short* __restrict__ Q,
    const unsigned short* __restrict__ K,
    const unsigned short* __restrict__ Vt,
    unsigned short* __restrict__ att) {
  __shared__ __align__(16) unsigned short ldsP[8 * 32 * LDSTRIDE];
  const int bh = blockIdx.x;
  const int b = bh >> 4, h = bh & 15;
  const unsigned short* qp = Q  + (size_t)bh * TT * DH;
  const unsigned short* kp = K  + (size_t)bh * TT * DH;
  const unsigned short* vp = Vt + (size_t)bh * DH * TT;
  const int tid = threadIdx.x, wave = tid >> 5, lane = tid & 31;
  const int ln = lane & 15, lh = lane >> 4;
  const int qbase = wave * 32;
  unsigned short* pbuf = ldsP + wave * 32 * LDSTRIDE;

  // Q A-fragments (held in registers across all key blocks)
  v16bf aQ[2][2];
#pragma unroll
  for (int mi = 0; mi < 2; ++mi)
#pragma unroll
    for (int kk = 0; kk < 2; ++kk) {
      const unsigned short* p =
          qp + (size_t)(qbase + mi * 16 + ln) * DH + kk * 32 + (lh << 4);
      BFrag16 f;
      f.h[0] = *(const v8bf*)(p);
      f.h[1] = *(const v8bf*)(p + 8);
      aQ[mi][kk] = f.v;
    }

  v8f zero = {};
  v8f O[2][4];
  float mrow[2][8], lrow[2][8];
#pragma unroll
  for (int mi = 0; mi < 2; ++mi) {
#pragma unroll
    for (int ni = 0; ni < 4; ++ni) O[mi][ni] = zero;
#pragma unroll
    for (int v = 0; v < 8; ++v) { mrow[mi][v] = -1e30f; lrow[mi][v] = 0.f; }
  }

  const int nblocks = wave + 1;  // causal: skip fully-masked key blocks
  for (int jb = 0; jb < nblocks; ++jb) {
    const int j0 = jb * 32;
    // prefetch next key block (K rows and V^T columns) into cache
    if (jb + 1 < nblocks) {
      __builtin_prefetch(kp + (size_t)(j0 + 32 + lane) * DH, 0, 0);
      __builtin_prefetch(vp + (size_t)lane * TT + j0 + 32, 0, 0);
      __builtin_prefetch(vp + (size_t)(lane + 32) * TT + j0 + 32, 0, 0);
    }
    // ---- S = Q * K^T ----
    v16bf bK[2][2];
#pragma unroll
    for (int ni = 0; ni < 2; ++ni)
#pragma unroll
      for (int kk = 0; kk < 2; ++kk) {
        // B elem (d,n) = K[n][d]: lane = key row, contiguous in d
        const unsigned short* p =
            kp + (size_t)(j0 + ni * 16 + ln) * DH + kk * 32 + (lh << 4);
        BFrag16 f;
        f.h[0] = *(const v8bf*)(p);
        f.h[1] = *(const v8bf*)(p + 8);
        bK[ni][kk] = f.v;
      }
    v8f S[2][2];
#pragma unroll
    for (int mi = 0; mi < 2; ++mi)
#pragma unroll
      for (int ni = 0; ni < 2; ++ni) {
        v8f s = wmma_bf16(aQ[mi][0], bK[ni][0], zero);
        S[mi][ni] = wmma_bf16(aQ[mi][1], bK[ni][1], s);
      }
    // ---- online softmax + P to LDS (bf16, transposed into A layout) ----
#pragma unroll
    for (int mi = 0; mi < 2; ++mi) {
#pragma unroll
      for (int v = 0; v < 8; ++v) {
        const int q  = qbase + mi * 16 + v + (lh << 3);
        const int c0 = j0 + ln, c1 = c0 + 16;
        float s0 = S[mi][0][v] * SCALE_F;
        float s1 = S[mi][1][v] * SCALE_F;
        if (c0 > q) s0 = -1e30f;
        if (c1 > q) s1 = -1e30f;
        float bm = fmaxf(s0, s1);
        bm = fmaxf(bm, __shfl_xor(bm, 1, 32));
        bm = fmaxf(bm, __shfl_xor(bm, 2, 32));
        bm = fmaxf(bm, __shfl_xor(bm, 4, 32));
        bm = fmaxf(bm, __shfl_xor(bm, 8, 32));
        const float mnew = fmaxf(mrow[mi][v], bm);
        const float corr = __expf(mrow[mi][v] - mnew);
        const float p0 = __expf(s0 - mnew);
        const float p1 = __expf(s1 - mnew);
        float rs = p0 + p1;
        rs += __shfl_xor(rs, 1, 32);
        rs += __shfl_xor(rs, 2, 32);
        rs += __shfl_xor(rs, 4, 32);
        rs += __shfl_xor(rs, 8, 32);
        lrow[mi][v] = lrow[mi][v] * corr + rs;
        mrow[mi][v] = mnew;
#pragma unroll
        for (int ni = 0; ni < 4; ++ni) O[mi][ni][v] *= corr;
        const int r = mi * 16 + v + (lh << 3);
        pbuf[r * LDSTRIDE + ln]      = f32_to_bf16(p0);
        pbuf[r * LDSTRIDE + 16 + ln] = f32_to_bf16(p1);
      }
    }
    asm volatile("s_wait_dscnt 0x0" ::: "memory");  // cross-lane LDS visibility
    // ---- O += P * V ----
    v16bf aP[2];
#pragma unroll
    for (int mi = 0; mi < 2; ++mi)
      aP[mi] = frag_a(pbuf + mi * 16 * LDSTRIDE, LDSTRIDE);
    v16bf bV[4];
#pragma unroll
    for (int ni = 0; ni < 4; ++ni) {
      // Vt[d][t]: lane = output column d, contiguous in key index
      const unsigned short* p =
          vp + (size_t)(ni * 16 + ln) * TT + j0 + (lh << 4);
      BFrag16 f;
      f.h[0] = *(const v8bf*)(p);
      f.h[1] = *(const v8bf*)(p + 8);
      bV[ni] = f.v;
    }
#pragma unroll
    for (int mi = 0; mi < 2; ++mi)
#pragma unroll
      for (int ni = 0; ni < 4; ++ni)
        O[mi][ni] = wmma_bf16(aP[mi], bV[ni], O[mi][ni]);
  }

  // ---- finalize: O / l, store merged-head layout [B*T][C] ----
#pragma unroll
  for (int mi = 0; mi < 2; ++mi)
#pragma unroll
    for (int v = 0; v < 8; ++v) {
      const float inv = 1.0f / lrow[mi][v];
      const int r = qbase + mi * 16 + v + (lh << 3);
      const size_t row = (size_t)b * TT + r;
#pragma unroll
      for (int ni = 0; ni < 4; ++ni) {
        const int col = h * DH + ni * 16 + ln;
        att[row * CC + col] = f32_to_bf16(O[mi][ni][v] * inv);
      }
    }
}

// ---------------- kernel 4: output projection + bias (fp32 out) ----------------
__global__ void __launch_bounds__(256) proj_gemm_kernel(
    const unsigned short* __restrict__ att,
    const unsigned short* __restrict__ wto,
    const float* __restrict__ bo,
    float* __restrict__ out) {
  __shared__ __align__(16) unsigned short ldsA[2 * BUFSH];
  __shared__ __align__(16) unsigned short ldsB[2 * BUFSH];
  const int tid = threadIdx.x;
  const int wave = tid >> 5;
  const int wm = wave >> 1;
  const int wn = wave & 1;
  const int M0 = blockIdx.x * 128;
  const int N0 = blockIdx.y * 128;
  const unsigned ldsAo = to_lds_off(ldsA);
  const unsigned ldsBo = to_lds_off(ldsB);

  v8f zero = {};
  v8f acc[2][4];
#pragma unroll
  for (int mi = 0; mi < 2; ++mi)
#pragma unroll
    for (int ni = 0; ni < 4; ++ni) acc[mi][ni] = zero;

  stage_tile_async(ldsAo, att, M0, 0);
  stage_tile_async(ldsBo, wto, N0, 0);

  for (int it = 0; it < CC / 32; ++it) {
    const int p = it & 1;
    if (it + 1 < CC / 32) {
      stage_tile_async(ldsAo + (1 - p) * BUFBYTES, att, M0, (it + 1) * 32);
      stage_tile_async(ldsBo + (1 - p) * BUFBYTES, wto, N0, (it + 1) * 32);
      asm volatile("s_wait_asynccnt 0x4" ::: "memory");
    } else {
      asm volatile("s_wait_asynccnt 0x0" ::: "memory");
    }
    __syncthreads();
    const unsigned short* bufA = ldsA + p * BUFSH;
    const unsigned short* bufB = ldsB + p * BUFSH;
    v16bf aF[2], bF[4];
#pragma unroll
    for (int mi = 0; mi < 2; ++mi)
      aF[mi] = frag_a(bufA + (wm * 32 + mi * 16) * LDSTRIDE, LDSTRIDE);
#pragma unroll
    for (int ni = 0; ni < 4; ++ni)
      bF[ni] = frag_b(bufB + (wn * 64 + ni * 16) * LDSTRIDE, LDSTRIDE);
#pragma unroll
    for (int mi = 0; mi < 2; ++mi)
#pragma unroll
      for (int ni = 0; ni < 4; ++ni)
        acc[mi][ni] = wmma_bf16(aF[mi], bF[ni], acc[mi][ni]);
    __syncthreads();
  }

  const int lane = tid & 31, ln = lane & 15, lh = lane >> 4;
#pragma unroll
  for (int mi = 0; mi < 2; ++mi)
#pragma unroll
    for (int ni = 0; ni < 4; ++ni) {
      const int gn = N0 + wn * 64 + ni * 16 + ln;
      const float bias = bo[gn];
#pragma unroll
      for (int v = 0; v < 8; ++v) {
        const int gm = M0 + wm * 32 + mi * 16 + v + (lh << 3);
        out[(size_t)gm * CC + gn] = acc[mi][ni][v] + bias;
      }
    }
}

// ---------------- host launcher ----------------
extern "C" void kernel_launch(void* const* d_in, const int* in_sizes, int n_in,
                              void* d_out, int out_size, void* d_ws, size_t ws_size,
                              hipStream_t stream) {
  const float* x  = (const float*)d_in[0];
  const float* Wq = (const float*)d_in[1];
  const float* Wk = (const float*)d_in[2];
  const float* Wv = (const float*)d_in[3];
  const float* Wo = (const float*)d_in[4];
  const float* bo = (const float*)d_in[5];
  float* out = (float*)d_out;

  char* ws = (char*)d_ws;
  const size_t XB = (size_t)MM * CC * 2;   // 32 MB (bf16 activations)
  const size_t WB = (size_t)CC * CC * 2;   // 2 MB  (bf16 weight)
  unsigned short* xb   = (unsigned short*)(ws);
  unsigned short* wtq  = (unsigned short*)(ws + XB);
  unsigned short* wtk  = (unsigned short*)(ws + XB + 1 * WB);
  unsigned short* wtv  = (unsigned short*)(ws + XB + 2 * WB);
  unsigned short* wto  = (unsigned short*)(ws + XB + 3 * WB);
  unsigned short* qws  = (unsigned short*)(ws + 1 * XB + 4 * WB);
  unsigned short* kws  = (unsigned short*)(ws + 2 * XB + 4 * WB);
  unsigned short* vtws = (unsigned short*)(ws + 3 * XB + 4 * WB);
  unsigned short* attw = (unsigned short*)(ws + 4 * XB + 4 * WB);

  cvt_bf16_kernel<<<2048, 256, 0, stream>>>(x, xb, MM * CC);
  transpose_bf16_kernel<<<4096, 256, 0, stream>>>(Wq, wtq);
  transpose_bf16_kernel<<<4096, 256, 0, stream>>>(Wk, wtk);
  transpose_bf16_kernel<<<4096, 256, 0, stream>>>(Wv, wtv);
  transpose_bf16_kernel<<<4096, 256, 0, stream>>>(Wo, wto);

  qkv_gemm_kernel<<<dim3(MM / 128, CC / 128, 3), 256, 0, stream>>>(
      xb, wtq, wtk, wtv, qws, kws, vtws);

  attention_kernel<<<BB * HH, 256, 0, stream>>>(qws, kws, vtws, attw);

  proj_gemm_kernel<<<dim3(MM / 128, CC / 128), 256, 0, stream>>>(
      attw, wto, bo, out);
}
